// S_CEMBlock_18743237280783
// MI455X (gfx1250) — compile-verified, matrix operand order checked by reading, not run on hardware
//
#include <hip/hip_runtime.h>
#include <hip/hip_bf16.h>
#include <math.h>

// ---------------- problem constants ----------------
#define BB   2
#define CC   48
#define HDS  3
#define CPH  16
#define HH   64
#define WW   64
#define NN   4096          // H*W
#define QKVC 144           // 3*C

typedef __attribute__((ext_vector_type(16))) _Float16 v16h;
typedef __attribute__((ext_vector_type(8)))  float    v8f;

// D = A(16x32 f16) * B(32x16 f16, stored column-per-lane == rows of B^T) + C
__device__ __forceinline__ v8f wmma16(v16h a, v16h b, v8f c) {
  return __builtin_amdgcn_wmma_f32_16x16x32_f16(
      /*neg_a=*/false, a, /*neg_b=*/false, b,
      /*c_mod=*/(short)0, c, /*reuse_a=*/false, /*reuse_b=*/false);
}

// A/B 16-bit fragment layout (ISA 7.12.2): lane L -> row m=L%16, hi=L/16,
// halfs[0..7]  = K = k0 + hi*8 + i
// halfs[8..15] = K = k0 + 16 + hi*8 + i
__device__ __forceinline__ v16h frag_from2(const _Float16* p_lo, const _Float16* p_hi) {
  v16h r;
#pragma unroll
  for (int i = 0; i < 8; ++i) { r[i] = p_lo[i]; r[i + 8] = p_hi[i]; }
  return r;
}
// K zero-padded variant (valid K=0..15 only, upper 16 K's are zero)
__device__ __forceinline__ v16h frag_lo(const _Float16* p_lo) {
  v16h r;
#pragma unroll
  for (int i = 0; i < 8; ++i) { r[i] = p_lo[i]; r[i + 8] = (_Float16)0.f; }
  return r;
}

// ---------------- K1: LayerNorm1 + 1x1 conv (48 -> 144) ----------------
__global__ void ln_qkv_kernel(const float* __restrict__ inp,
                              const float* __restrict__ ln_w, const float* __restrict__ ln_b,
                              const float* __restrict__ qkv_w, const float* __restrict__ qkv_b,
                              float* __restrict__ qkv_pre) {
  __shared__ float w[QKVC * CC];
  __shared__ float bia[QKVC];
  for (int i = threadIdx.x; i < QKVC * CC; i += blockDim.x) w[i] = qkv_w[i];
  for (int i = threadIdx.x; i < QKVC; i += blockDim.x) bia[i] = qkv_b[i];
  __syncthreads();
  int idx = blockIdx.x * blockDim.x + threadIdx.x;     // (b, n)
  if (idx >= BB * NN) return;
  int b = idx >> 12, n = idx & (NN - 1);
  const float* xp = inp + (size_t)b * CC * NN + n;
  float x[CC];
  float mu = 0.f;
#pragma unroll
  for (int c = 0; c < CC; ++c) { x[c] = xp[(size_t)c * NN]; mu += x[c]; }
  mu *= (1.f / CC);
  float var = 0.f;
#pragma unroll
  for (int c = 0; c < CC; ++c) { float d = x[c] - mu; var += d * d; }
  var *= (1.f / CC);
  float inv = rsqrtf(var + 1e-6f);
#pragma unroll
  for (int c = 0; c < CC; ++c) x[c] = ln_w[c] * ((x[c] - mu) * inv) + ln_b[c];
  float* op = qkv_pre + (size_t)b * QKVC * NN + n;
  for (int o = 0; o < QKVC; ++o) {
    float a = bia[o];
    const float* wr = &w[o * CC];
#pragma unroll
    for (int c = 0; c < CC; ++c) a += wr[c] * x[c];
    op[(size_t)o * NN] = a;
  }
}

// ---------------- K2: depthwise 3x3, SAME ----------------
__global__ void dwconv_kernel(const float* __restrict__ src,
                              const float* __restrict__ w, const float* __restrict__ bias,
                              float* __restrict__ dst) {
  int idx = blockIdx.x * blockDim.x + threadIdx.x;     // (b*144+ch, n)
  if (idx >= BB * QKVC * NN) return;
  int n = idx & (NN - 1);
  int bc = idx >> 12;
  int ch = bc % QKVC;
  int x = n & (WW - 1), y = n >> 6;
  const float* sp = src + (size_t)bc * NN;
  const float* wp = w + ch * 9;
  float acc = bias[ch];
#pragma unroll
  for (int dy = -1; dy <= 1; ++dy) {
    int yy = y + dy;
    if (yy < 0 || yy >= HH) continue;
#pragma unroll
    for (int dx = -1; dx <= 1; ++dx) {
      int xx = x + dx;
      if (xx < 0 || xx >= WW) continue;
      acc += wp[(dy + 1) * 3 + (dx + 1)] * sp[yy * WW + xx];
    }
  }
  dst[(size_t)bc * NN + n] = acc;
}

// ---------------- K3: channel-attention row inv-norms over n ----------------
// qkv channels 0..47 = q, 48..95 = k.  invnorm[b*96 + j]
__global__ void rownorm_kernel(const float* __restrict__ qkv, float* __restrict__ invnorm) {
  int b = blockIdx.x / 96;
  int j = blockIdx.x % 96;
  const float* p = qkv + ((size_t)b * QKVC + j) * NN;
  __shared__ float red[256];
  float s = 0.f;
  for (int i = threadIdx.x; i < NN; i += 256) { float v = p[i]; s += v * v; }
  red[threadIdx.x] = s;
  __syncthreads();
  for (int st = 128; st > 0; st >>= 1) {
    if (threadIdx.x < st) red[threadIdx.x] += red[threadIdx.x + st];
    __syncthreads();
  }
  if (threadIdx.x == 0) invnorm[b * 96 + j] = 1.f / fmaxf(sqrtf(red[0]), 1e-12f);
}

// ---------------- K4: build f16 operand tensors in WMMA-friendly layouts ----
// qc,kc,v16 : [bh][16][4096]   (row c, contiguous n)   -- channel attn / P@V
// qs,ks,vT  : [bh][4096][16]   (row n, contiguous c)   -- spatial attn / attn_c@V
__global__ void prep_kernel(const float* __restrict__ qkv, const float* __restrict__ invnorm,
                            _Float16* __restrict__ qc, _Float16* __restrict__ kc,
                            _Float16* __restrict__ qs, _Float16* __restrict__ ks,
                            _Float16* __restrict__ v16, _Float16* __restrict__ vT) {
  int idx = blockIdx.x * blockDim.x + threadIdx.x;     // (bh, n)
  if (idx >= BB * HDS * NN) return;
  int n = idx & (NN - 1);
  int bh = idx >> 12;
  int b = bh / HDS, h = bh % HDS;
  const float* base = qkv + (size_t)b * QKVC * NN;
  float qv[CPH], kv[CPH], vv[CPH];
  float qss = 0.f, kss = 0.f;
#pragma unroll
  for (int d = 0; d < CPH; ++d) {
    int c = h * CPH + d;
    float q = base[(size_t)c * NN + n];
    float k = base[(size_t)(CC + c) * NN + n];
    float v = base[(size_t)(2 * CC + c) * NN + n];
    qv[d] = q; kv[d] = k; vv[d] = v;
    qss += q * q; kss += k * k;
  }
  float qinv = 1.f / fmaxf(sqrtf(qss), 1e-12f);   // spatial: norm over cph
  float kinv = 1.f / fmaxf(sqrtf(kss), 1e-12f);
  size_t tb = (size_t)bh * NN * CPH + (size_t)n * CPH;
  size_t cb = (size_t)bh * CPH * NN;
#pragma unroll
  for (int d = 0; d < CPH; ++d) {
    int c = h * CPH + d;
    qs[tb + d] = (_Float16)(qv[d] * qinv);
    ks[tb + d] = (_Float16)(kv[d] * kinv);
    vT[tb + d] = (_Float16)vv[d];
    float qrn = invnorm[b * 96 + c];              // channel: norm over n
    float krn = invnorm[b * 96 + CC + c];
    qc[cb + (size_t)d * NN + n]  = (_Float16)(qv[d] * qrn);
    kc[cb + (size_t)d * NN + n]  = (_Float16)(kv[d] * krn);
    v16[cb + (size_t)d * NN + n] = (_Float16)vv[d];
  }
}

// ---------------- K5: channel attention (one wave per (b,h)) ----------------
// S = Qc*Kc^T (16x16, K=4096 via 128 WMMAs) -> relu/softmax -> outc = P*V
__global__ void chan_attn_kernel(const _Float16* __restrict__ qc, const _Float16* __restrict__ kc,
                                 const _Float16* __restrict__ vT, const float* __restrict__ temp1,
                                 float* __restrict__ outc) {
  int bh = blockIdx.x;                 // 0..5
  int h = bh % HDS;
  int lane = threadIdx.x;
  int m = lane & 15, hi = lane >> 4;
  const _Float16* qb = qc + (size_t)bh * CPH * NN;
  const _Float16* kb = kc + (size_t)bh * CPH * NN;
  const _Float16* vb = vT + (size_t)bh * NN * CPH;
  __shared__ float    sS[CPH][CPH];
  __shared__ _Float16 sP[CPH][CPH];
  v8f acc = {};
  for (int k0 = 0; k0 < NN; k0 += 32) {
    v16h a = frag_from2(qb + m * NN + k0 + hi * 8, qb + m * NN + k0 + 16 + hi * 8);
    v16h b = frag_from2(kb + m * NN + k0 + hi * 8, kb + m * NN + k0 + 16 + hi * 8);
    acc = wmma16(a, b, acc);
  }
  float t = temp1[h];
#pragma unroll
  for (int r = 0; r < 8; ++r) sS[r + 8 * hi][m] = fmaxf(acc[r] * t, 0.f);   // relu
  __syncthreads();
  if (lane < CPH) {                                     // row softmax
    float e[CPH], sum = 0.f;
#pragma unroll
    for (int c = 0; c < CPH; ++c) { e[c] = __expf(sS[lane][c]); sum += e[c]; }
    float inv = 1.f / sum;
#pragma unroll
    for (int c = 0; c < CPH; ++c) sP[lane][c] = (_Float16)(e[c] * inv);
  }
  __syncthreads();
  v16h pA = frag_lo(&sP[m][hi * 8]);                    // K = 16 (padded to 32)
  float* ob = outc + (size_t)bh * CPH * NN;
  for (int n0 = 0; n0 < NN; n0 += 16) {
    v16h bV = frag_lo(vb + (size_t)(n0 + m) * CPH + hi * 8);
    v8f o = {};
    o = wmma16(pA, bV, o);
#pragma unroll
    for (int r = 0; r < 8; ++r) ob[(size_t)(r + 8 * hi) * NN + n0 + m] = o[r];
  }
}

// ---------------- K6: spatial attention (one wave per 16-row q tile) -------
// relu clamps scores to [0, temp2] before softmax -> exp is bounded, so a
// single-pass softmax (no running max) is numerically safe: accumulate
// num = sum exp(s)*v via WMMA and den = sum exp(s) per row, divide at end.
__global__ void spat_attn_kernel(const _Float16* __restrict__ qs, const _Float16* __restrict__ ks,
                                 const _Float16* __restrict__ v16, const float* __restrict__ temp2,
                                 float* __restrict__ outs) {
  int blk = blockIdx.x;
  int bh = blk >> 8;                   // 256 tiles per (b,h)
  int h = bh % HDS;
  int n0 = (blk & 255) * 16;
  int lane = threadIdx.x;
  int m = lane & 15, hi = lane >> 4;
  const _Float16* qbase = qs + (size_t)bh * NN * CPH;
  const _Float16* kbase = ks + (size_t)bh * NN * CPH;
  const _Float16* vbase = v16 + (size_t)bh * CPH * NN;
  __shared__ float sP[CPH][32];
  __shared__ float sden[32];
  __shared__ float sdrow[CPH];
  v16h aQ = frag_lo(qbase + (size_t)(n0 + m) * CPH + hi * 8);   // K=cph=16 padded
  float t = temp2[h];
  v8f O = {};
  float den = 0.f;
  for (int m0 = 0; m0 < NN; m0 += 32) {
    v16h bk1 = frag_lo(kbase + (size_t)(m0 + m) * CPH + hi * 8);
    v16h bk2 = frag_lo(kbase + (size_t)(m0 + 16 + m) * CPH + hi * 8);
    v8f z = {};
    v8f S1 = wmma16(aQ, bk1, z);       // keys m0..m0+15
    v8f S2 = wmma16(aQ, bk2, z);       // keys m0+16..m0+31
#pragma unroll
    for (int r = 0; r < 8; ++r) {
      sP[r + 8 * hi][m]      = __expf(fmaxf(S1[r] * t, 0.f));
      sP[r + 8 * hi][16 + m] = __expf(fmaxf(S2[r] * t, 0.f));
    }
    __syncthreads();
    v16h pA;                            // repack P into A-fragment (rows=q, K=keys)
#pragma unroll
    for (int i = 0; i < 8; ++i) {
      float plo = sP[m][hi * 8 + i];
      float phi = sP[m][16 + hi * 8 + i];
      den += plo + phi;
      pA[i]     = (_Float16)plo;
      pA[i + 8] = (_Float16)phi;
    }
    v16h bV = frag_from2(vbase + (size_t)m * NN + m0 + hi * 8,
                         vbase + (size_t)m * NN + m0 + 16 + hi * 8);
    O = wmma16(pA, bV, O);
    __syncthreads();
  }
  sden[lane] = den;
  __syncthreads();
  if (lane < CPH) sdrow[lane] = sden[lane] + sden[lane + 16];   // both K-halves of row
  __syncthreads();
  float* ob = outs + (size_t)bh * CPH * NN;
#pragma unroll
  for (int r = 0; r < 8; ++r) {
    int row = r + 8 * hi;              // q row within tile; D col = channel m
    ob[(size_t)m * NN + n0 + row] = O[r] / sdrow[row];
  }
}

// ---------------- K7: po1/po2 projections + residual -----------------------
__global__ void proj_res_kernel(const float* __restrict__ inp,
                                const float* __restrict__ outc, const float* __restrict__ outs,
                                const float* __restrict__ po1_w, const float* __restrict__ po1_b,
                                const float* __restrict__ po2_w, const float* __restrict__ po2_b,
                                const float* __restrict__ beta, const float* __restrict__ beta2,
                                float* __restrict__ y) {
  __shared__ float w1[CC * CC], w2[CC * CC];
  for (int i = threadIdx.x; i < CC * CC; i += blockDim.x) { w1[i] = po1_w[i]; w2[i] = po2_w[i]; }
  __syncthreads();
  int idx = blockIdx.x * blockDim.x + threadIdx.x;
  if (idx >= BB * NN) return;
  int b = idx >> 12, n = idx & (NN - 1);
  float oc[CC], os[CC];
  const float* pc = outc + (size_t)b * CC * NN + n;
  const float* ps = outs + (size_t)b * CC * NN + n;
#pragma unroll
  for (int c = 0; c < CC; ++c) { oc[c] = pc[(size_t)c * NN]; os[c] = ps[(size_t)c * NN]; }
  const float* pi = inp + (size_t)b * CC * NN + n;
  float* py = y + (size_t)b * CC * NN + n;
  for (int c = 0; c < CC; ++c) {
    float a1 = po1_b[c], a2 = po2_b[c];
    const float* r1 = &w1[c * CC];
    const float* r2 = &w2[c * CC];
#pragma unroll
    for (int d = 0; d < CC; ++d) { a1 += r1[d] * oc[d]; a2 += r2[d] * os[d]; }
    py[(size_t)c * NN] = pi[(size_t)c * NN] + a1 * beta[c] + a2 * beta2[c];
  }
}

// ---------------- K8: LN2 -> conv4 -> SimpleGate -> conv5 -> residual ------
__global__ void ffn_kernel(const float* __restrict__ y,
                           const float* __restrict__ ln_w, const float* __restrict__ ln_b,
                           const float* __restrict__ w4, const float* __restrict__ b4,
                           const float* __restrict__ w5, const float* __restrict__ b5,
                           const float* __restrict__ gamma, float* __restrict__ out) {
  __shared__ float sw4[2 * CC * CC], sw5[CC * CC];
  for (int i = threadIdx.x; i < 2 * CC * CC; i += blockDim.x) sw4[i] = w4[i];
  for (int i = threadIdx.x; i < CC * CC; i += blockDim.x) sw5[i] = w5[i];
  __syncthreads();
  int idx = blockIdx.x * blockDim.x + threadIdx.x;
  if (idx >= BB * NN) return;
  int b = idx >> 12, n = idx & (NN - 1);
  const float* yp = y + (size_t)b * CC * NN + n;
  float yv[CC], x[CC];
  float mu = 0.f;
#pragma unroll
  for (int c = 0; c < CC; ++c) { yv[c] = yp[(size_t)c * NN]; mu += yv[c]; }
  mu *= (1.f / CC);
  float var = 0.f;
#pragma unroll
  for (int c = 0; c < CC; ++c) { float d = yv[c] - mu; var += d * d; }
  var *= (1.f / CC);
  float inv = rsqrtf(var + 1e-6f);
#pragma unroll
  for (int c = 0; c < CC; ++c) x[c] = ln_w[c] * ((yv[c] - mu) * inv) + ln_b[c];
  float g[CC];
  for (int o = 0; o < CC; ++o) {
    float a1 = b4[o], a2 = b4[o + CC];
    const float* r1 = &sw4[o * CC];
    const float* r2 = &sw4[(o + CC) * CC];
#pragma unroll
    for (int c = 0; c < CC; ++c) { a1 += r1[c] * x[c]; a2 += r2[c] * x[c]; }
    g[o] = a1 * a2;                                   // SimpleGate
  }
  float* po = out + (size_t)b * CC * NN + n;
  for (int c = 0; c < CC; ++c) {
    float a = b5[c];
    const float* r = &sw5[c * CC];
#pragma unroll
    for (int d = 0; d < CC; ++d) a += r[d] * g[d];
    po[(size_t)c * NN] = yv[c] + a * gamma[c];
  }
}

// ---------------- launch ----------------------------------------------------
extern "C" void kernel_launch(void* const* d_in, const int* in_sizes, int n_in,
                              void* d_out, int out_size, void* d_ws, size_t ws_size,
                              hipStream_t stream) {
  const float* inp     = (const float*)d_in[0];
  const float* qkv_w   = (const float*)d_in[1];
  const float* qkv_b   = (const float*)d_in[2];
  const float* dw_w    = (const float*)d_in[3];
  const float* dw_b    = (const float*)d_in[4];
  const float* po1_w   = (const float*)d_in[5];
  const float* po1_b   = (const float*)d_in[6];
  const float* po2_w   = (const float*)d_in[7];
  const float* po2_b   = (const float*)d_in[8];
  const float* temp1   = (const float*)d_in[9];
  const float* temp2   = (const float*)d_in[10];
  const float* conv4_w = (const float*)d_in[11];
  const float* conv4_b = (const float*)d_in[12];
  const float* conv5_w = (const float*)d_in[13];
  const float* conv5_b = (const float*)d_in[14];
  const float* ln1_w   = (const float*)d_in[15];
  const float* ln1_b   = (const float*)d_in[16];
  const float* ln2_w   = (const float*)d_in[17];
  const float* ln2_b   = (const float*)d_in[18];
  const float* beta    = (const float*)d_in[19];
  const float* beta2   = (const float*)d_in[20];
  const float* gamma   = (const float*)d_in[21];

  char* ws = (char*)d_ws;
  size_t off = 0;
  auto alloc = [&](size_t bytes) { char* p = ws + off; off = (off + bytes + 255) & ~(size_t)255; return p; };

  float*    qkv_pre = (float*)   alloc((size_t)BB * QKVC * NN * 4);
  float*    qkv     = (float*)   alloc((size_t)BB * QKVC * NN * 4);
  float*    invnorm = (float*)   alloc((size_t)BB * 96 * 4);
  _Float16* qc      = (_Float16*)alloc((size_t)BB * CC * NN * 2);
  _Float16* kc      = (_Float16*)alloc((size_t)BB * CC * NN * 2);
  _Float16* qs      = (_Float16*)alloc((size_t)BB * CC * NN * 2);
  _Float16* ks      = (_Float16*)alloc((size_t)BB * CC * NN * 2);
  _Float16* v16     = (_Float16*)alloc((size_t)BB * CC * NN * 2);
  _Float16* vT      = (_Float16*)alloc((size_t)BB * CC * NN * 2);
  float*    outc    = (float*)   alloc((size_t)BB * CC * NN * 4);
  float*    outs    = (float*)   alloc((size_t)BB * CC * NN * 4);
  float*    ybuf    = (float*)   alloc((size_t)BB * CC * NN * 4);
  (void)ws_size; (void)in_sizes; (void)n_in; (void)out_size;

  ln_qkv_kernel<<<(BB * NN) / 256, 256, 0, stream>>>(inp, ln1_w, ln1_b, qkv_w, qkv_b, qkv_pre);
  dwconv_kernel<<<(BB * QKVC * NN) / 256, 256, 0, stream>>>(qkv_pre, dw_w, dw_b, qkv);
  rownorm_kernel<<<BB * 96, 256, 0, stream>>>(qkv, invnorm);
  prep_kernel<<<(BB * HDS * NN) / 256, 256, 0, stream>>>(qkv, invnorm, qc, kc, qs, ks, v16, vT);
  chan_attn_kernel<<<BB * HDS, 32, 0, stream>>>(qc, kc, vT, temp1, outc);
  spat_attn_kernel<<<BB * HDS * 256, 32, 0, stream>>>(qs, ks, v16, temp2, outs);
  proj_res_kernel<<<(BB * NN) / 256, 256, 0, stream>>>(inp, outc, outs, po1_w, po1_b, po2_w, po2_b,
                                                       beta, beta2, ybuf);
  ffn_kernel<<<(BB * NN) / 256, 256, 0, stream>>>(ybuf, ln2_w, ln2_b, conv4_w, conv4_b,
                                                  conv5_w, conv5_b, gamma, (float*)d_out);
}
